// cross_NonLocalBlock_12249246728627
// MI455X (gfx1250) — matrix operand order repaired and checked by reading, not compiled, not run on hardware
//
#include <hip/hip_runtime.h>

// ---------------- problem constants ----------------
#define BN_  8
#define C_   1024
#define IC_  512
#define N_   1024
#define NL_  4096
#define M_   5120   // N_ + NL_

typedef __bf16 v16bf __attribute__((ext_vector_type(16)));
typedef __bf16 v8bf  __attribute__((ext_vector_type(8)));
typedef float  v8f   __attribute__((ext_vector_type(8)));
typedef unsigned int uint4v __attribute__((ext_vector_type(4)));
typedef int int8v  __attribute__((ext_vector_type(8)));
typedef int int4v  __attribute__((ext_vector_type(4)));

#if __has_builtin(__builtin_amdgcn_tensor_load_to_lds)
#define HAVE_TDM 1
#else
#define HAVE_TDM 0
#endif

// fp32 -> bf16 round-to-nearest-even
__device__ __forceinline__ __bf16 f2bf(float f) {
  union { float f; unsigned u; } v; v.f = f;
  unsigned r = v.u + 0x7FFFu + ((v.u >> 16) & 1u);
  union { unsigned short s; __bf16 b; } o; o.s = (unsigned short)(r >> 16);
  return o.b;
}

__device__ __forceinline__ v8f wmma_bf16(v16bf a, v16bf b, v8f c) {
  return __builtin_amdgcn_wmma_f32_16x16x32_bf16(false, a, false, b, (short)0, c, false, false);
}

// A fragment (16x32 bf16): lane<16 -> row=lane, K {0..7,16..23}; lane>=16 -> row=lane-16, K {8..15,24..31}
__device__ __forceinline__ v16bf load_frag_a(const __bf16* base, int ld, int lane) {
  int r  = lane & 15;
  int ko = (lane < 16) ? 0 : 8;
  const __bf16* p = base + (long long)r * ld + ko;
  v8bf lo = *(const v8bf*)(p);
  v8bf hi = *(const v8bf*)(p + 16);
  return __builtin_shufflevector(lo, hi, 0,1,2,3,4,5,6,7,8,9,10,11,12,13,14,15);
}

// B fragment (32x16 bf16), operand stored col-major (o-major): elem(col j, k) = base[j*ld + k]
__device__ __forceinline__ v16bf load_frag_b(const __bf16* base, int ld, int lane) {
  int j  = lane & 15;
  int ko = (lane < 16) ? 0 : 8;
  const __bf16* p = base + (long long)j * ld + ko;
  v8bf lo = *(const v8bf*)(p);
  v8bf hi = *(const v8bf*)(p + 16);
  return __builtin_shufflevector(lo, hi, 0,1,2,3,4,5,6,7,8,9,10,11,12,13,14,15);
}

#if HAVE_TDM
// Issue a 2D TDM load: tile_h rows x tile_w bf16 elems (row stride = stride_elems)
// from global `g` into LDS byte offset `lds_off` (packed row-major tile_h x tile_w).
// D# layout per CDNA5 ISA 8.3/8.4 (group0 128b, group1 256b; groups 2/3 zero -> 2D).
// This toolchain's builtin takes 6 args: (g0 u32x4, g1 i32x8, g2 i32x4, g3 i32x4, i32x8, cpol).
__device__ __forceinline__ void tdm_load_2d(unsigned lds_off, const void* g,
                                            unsigned tile_w, unsigned tile_h,
                                            unsigned long long stride_elems,
                                            unsigned tens_w, unsigned tens_h) {
  unsigned long long ga = (unsigned long long)(size_t)g;
  uint4v g0;
  g0[0] = 1u;                                            // count=1 (valid), user mode
  g0[1] = lds_off;                                       // lds_addr (bytes)
  g0[2] = (unsigned)(ga & 0xFFFFFFFFu);                  // global_addr[31:0]
  g0[3] = (unsigned)((ga >> 32) & 0x01FFFFFFu) | (2u << 30); // global_addr[56:32] | type=2
  int8v g1;
  g1[0] = (int)(1u << 16);                               // wg_mask=0 | data_size=1 (2B)
  g1[1] = (int)((tens_w & 0xFFFFu) << 16);               // tensor_dim0[15:0]
  g1[2] = (int)(((tens_w >> 16) & 0xFFFFu) | ((tens_h & 0xFFFFu) << 16)); // dim0 hi | dim1 lo
  g1[3] = (int)(((tens_h >> 16) & 0xFFFFu) | ((tile_w & 0xFFFFu) << 16)); // dim1 hi | tile_dim0
  g1[4] = (int)(tile_h & 0xFFFFu);                       // tile_dim1 | tile_dim2=0
  g1[5] = (int)(unsigned)(stride_elems & 0xFFFFFFFFull); // tensor_dim0_stride[31:0]
  g1[6] = (int)(unsigned)((stride_elems >> 32) & 0xFFFFull); // stride[47:32] | dim1_stride lo=0
  g1[7] = 0;
  int4v z4; z4[0]=0; z4[1]=0; z4[2]=0; z4[3]=0;
  int8v z8; z8[0]=0; z8[1]=0; z8[2]=0; z8[3]=0; z8[4]=0; z8[5]=0; z8[6]=0; z8[7]=0;
  __builtin_amdgcn_tensor_load_to_lds(g0, g1, z4, z4, z8, 0);
}
#endif

// ---------------- conversion kernels ----------------
__global__ __launch_bounds__(256) void transpose_cvt_kernel(const float* __restrict__ in,
                                                            __bf16* __restrict__ out,
                                                            int Ch, int Ntok) {
  long long i = (long long)blockIdx.x * 256 + threadIdx.x;
  int c = (int)(i % Ch);
  long long t = i / Ch;
  int n = (int)(t % Ntok);
  int b = (int)(t / Ntok);
  out[i] = f2bf(in[((long long)b * Ch + c) * Ntok + n]);
}

__global__ __launch_bounds__(256) void cvt_kernel(const float* __restrict__ in,
                                                  __bf16* __restrict__ out, int n) {
  int i = blockIdx.x * 256 + threadIdx.x;
  if (i < n) out[i] = f2bf(in[i]);
}

// ---------------- generic WMMA GEMM, TDM double-buffered ----------------
// out(M x Ncols) = A(M x K) @ B(K x Ncols) + bias ; B stored o-major (Ncols rows of K)
// mode 0: token-major bf16; mode 1: feature-major bf16; mode 2: feature-major f32
#define TLA 512    // A tile elems (16x32)
#define TLB 2048   // B tile elems (64x32)
__global__ __launch_bounds__(128) void gemm_bf16_kernel(
    const __bf16* __restrict__ A, int lda, long long strideA,
    const __bf16* __restrict__ Bm, int ldb,
    const float* __restrict__ bias,
    void* __restrict__ Out, int ldo, long long strideO,
    int K, int mode) {
  int lane = threadIdx.x & 31;
  int wave = threadIdx.x >> 5;
  int b = blockIdx.z;
  int mBase = blockIdx.x * 16;
  int colBase = blockIdx.y * 256 + wave * 64;

  const __bf16* Ab = A + (long long)b * strideA + (long long)mBase * lda;
  const __bf16* Bb = Bm + (long long)colBase * ldb;

  v8f zero = {0.f,0.f,0.f,0.f,0.f,0.f,0.f,0.f};
  v8f acc[4] = {zero, zero, zero, zero};

#if HAVE_TDM
  // per-wave private double-buffered staging: no barriers in the k-loop
  __shared__ __align__(16) __bf16 smem[4 * 2 * (TLA + TLB)];
  __bf16* myA = smem + wave * 2 * (TLA + TLB);
  __bf16* myB = myA + 2 * TLA;
  unsigned aoff = (unsigned)(size_t)(void*)myA;   // LDS byte offset (low 32 bits of aperture addr)
  unsigned boff = (unsigned)(size_t)(void*)myB;

  tdm_load_2d(aoff, Ab, 32, 16, (unsigned long long)lda, (unsigned)lda, 16u);
  tdm_load_2d(boff, Bb, 32, 64, (unsigned long long)ldb, (unsigned)ldb, 64u);

  for (int k0 = 0; k0 < K; k0 += 32) {
    int cur = (k0 >> 5) & 1;
    int nxt = cur ^ 1;
    if (k0 + 32 < K) {
      tdm_load_2d(aoff + nxt * (TLA * 2), Ab + k0 + 32, 32, 16,
                  (unsigned long long)lda, (unsigned)lda, 16u);
      tdm_load_2d(boff + nxt * (TLB * 2), Bb + k0 + 32, 32, 64,
                  (unsigned long long)ldb, (unsigned)ldb, 64u);
      __builtin_amdgcn_s_wait_tensorcnt(2);   // current pair done; next pair in flight
    } else {
      __builtin_amdgcn_s_wait_tensorcnt(0);
    }
    const __bf16* At = myA + cur * TLA;
    const __bf16* Bt = myB + cur * TLB;
    v16bf af  = load_frag_a(At, 32, lane);
    v16bf bf0 = load_frag_b(Bt + 0 * 512, 32, lane);
    v16bf bf1 = load_frag_b(Bt + 1 * 512, 32, lane);
    v16bf bf2 = load_frag_b(Bt + 2 * 512, 32, lane);
    v16bf bf3 = load_frag_b(Bt + 3 * 512, 32, lane);
    acc[0] = wmma_bf16(af, bf0, acc[0]);
    acc[1] = wmma_bf16(af, bf1, acc[1]);
    acc[2] = wmma_bf16(af, bf2, acc[2]);
    acc[3] = wmma_bf16(af, bf3, acc[3]);
  }
#else
  for (int k0 = 0; k0 < K; k0 += 32) {
    v16bf af = load_frag_a(Ab + k0, lda, lane);
    #pragma unroll
    for (int t = 0; t < 4; ++t) {
      v16bf bf = load_frag_b(Bb + (long long)(t * 16) * ldb + k0, ldb, lane);
      acc[t] = wmma_bf16(af, bf, acc[t]);
    }
  }
#endif

  int colL = lane & 15;
  int rowL = (lane >> 4) * 8;
  if (mode == 0) {
    __bf16* ob = (__bf16*)Out + (long long)b * strideO;
    #pragma unroll
    for (int t = 0; t < 4; ++t) {
      int col = colBase + t * 16 + colL;
      float bv = bias[col];
      #pragma unroll
      for (int v = 0; v < 8; ++v)
        ob[(long long)(mBase + rowL + v) * ldo + col] = f2bf(acc[t][v] + bv);
    }
  } else if (mode == 1) {
    __bf16* ob = (__bf16*)Out + (long long)b * strideO;
    #pragma unroll
    for (int t = 0; t < 4; ++t) {
      int col = colBase + t * 16 + colL;
      float bv = bias[col];
      #pragma unroll
      for (int v = 0; v < 8; ++v)
        ob[(long long)col * ldo + (mBase + rowL + v)] = f2bf(acc[t][v] + bv);
    }
  } else {
    float* of = (float*)Out + (long long)b * strideO;
    #pragma unroll
    for (int t = 0; t < 4; ++t) {
      int col = colBase + t * 16 + colL;
      float bv = bias[col];
      #pragma unroll
      for (int v = 0; v < 8; ++v)
        of[(long long)col * ldo + (mBase + rowL + v)] = acc[t][v] + bv;
    }
  }
}

// ---------------- fused flash attention ----------------
// Q: (B, N, IC) token-major; Kt: (B, M, IC) token-major; Vt: (B, IC, M) feature-major
#define ICPW 128   // IC slice per wave (4 waves)
__global__ __launch_bounds__(128) void attn_kernel(
    const __bf16* __restrict__ Q, const __bf16* __restrict__ Kt,
    const __bf16* __restrict__ Vt, __bf16* __restrict__ Y) {
  __shared__ float Sred[4][2][16][16];
  __shared__ __align__(16) __bf16 Prow[16][32];
  __shared__ float rowAlpha[16];
  __shared__ float rowScale[16];

  int lane = threadIdx.x & 31;
  int wave = threadIdx.x >> 5;
  int b = blockIdx.y;
  int nBase = blockIdx.x * 16;

  const __bf16* Qb = Q  + ((long long)b * N_ + nBase) * IC_ + wave * ICPW;
  const __bf16* Kb = Kt + (long long)b * M_ * IC_ + wave * ICPW;
  const __bf16* Vb = Vt + ((long long)b * IC_ + wave * ICPW) * M_;

  v16bf qf[4];
  #pragma unroll
  for (int kk = 0; kk < 4; ++kk) qf[kk] = load_frag_a(Qb + kk * 32, IC_, lane);

  v8f zero = {0.f,0.f,0.f,0.f,0.f,0.f,0.f,0.f};
  v8f o[8] = {zero, zero, zero, zero, zero, zero, zero, zero};
  float mrow = -3.0e38f, lrow = 0.0f;

  for (int ms = 0; ms < M_; ms += 32) {
    if (ms + 32 < M_) {
      __builtin_prefetch(Kb + (long long)(ms + 32) * IC_, 0, 1);
      __builtin_prefetch(Vb + (ms + 32), 0, 1);
    }
    v8f s0 = zero, s1 = zero;
    #pragma unroll
    for (int kk = 0; kk < 4; ++kk) {
      v16bf k0f = load_frag_b(Kb + (long long)ms * IC_ + kk * 32, IC_, lane);
      v16bf k1f = load_frag_b(Kb + (long long)(ms + 16) * IC_ + kk * 32, IC_, lane);
      s0 = wmma_bf16(qf[kk], k0f, s0);
      s1 = wmma_bf16(qf[kk], k1f, s1);
    }
    __syncthreads();
    {
      int r = (lane >> 4) * 8, c = lane & 15;
      #pragma unroll
      for (int v = 0; v < 8; ++v) { Sred[wave][0][r + v][c] = s0[v]; Sred[wave][1][r + v][c] = s1[v]; }
    }
    __syncthreads();
    if (wave == 0 && lane < 16) {
      float sv[32];
      #pragma unroll
      for (int j = 0; j < 16; ++j) {
        sv[j]      = Sred[0][0][lane][j] + Sred[1][0][lane][j] + Sred[2][0][lane][j] + Sred[3][0][lane][j];
        sv[16 + j] = Sred[0][1][lane][j] + Sred[1][1][lane][j] + Sred[2][1][lane][j] + Sred[3][1][lane][j];
      }
      float mx = mrow;
      #pragma unroll
      for (int j = 0; j < 32; ++j) mx = fmaxf(mx, sv[j]);
      float alpha = __expf(mrow - mx);
      float sum = 0.f;
      #pragma unroll
      for (int j = 0; j < 32; ++j) { float e = __expf(sv[j] - mx); sum += e; Prow[lane][j] = f2bf(e); }
      lrow = lrow * alpha + sum;
      mrow = mx;
      rowAlpha[lane] = alpha;
    }
    __syncthreads();
    float al[8];
    #pragma unroll
    for (int v = 0; v < 8; ++v) al[v] = rowAlpha[(lane >> 4) * 8 + v];
    v16bf pf = load_frag_a(&Prow[0][0], 32, lane);
    #pragma unroll
    for (int f = 0; f < 8; ++f) {
      #pragma unroll
      for (int v = 0; v < 8; ++v) o[f][v] *= al[v];
      v16bf vf = load_frag_b(Vb + (long long)(f * 16) * M_ + ms, M_, lane);
      o[f] = wmma_bf16(pf, vf, o[f]);
    }
  }

  __syncthreads();
  if (wave == 0 && lane < 16) rowScale[lane] = 1.0f / lrow;
  __syncthreads();
  float sc[8];
  #pragma unroll
  for (int v = 0; v < 8; ++v) sc[v] = rowScale[(lane >> 4) * 8 + v];
  __bf16* Yb = Y + ((long long)b * N_ + nBase) * IC_ + wave * ICPW;
  int rowL = (lane >> 4) * 8, colL = lane & 15;
  #pragma unroll
  for (int f = 0; f < 8; ++f)
    #pragma unroll
    for (int v = 0; v < 8; ++v)
      Yb[(long long)(rowL + v) * IC_ + f * 16 + colL] = f2bf(o[f][v] * sc[v]);
}

// ---------------- batchnorm ----------------
__global__ __launch_bounds__(256) void bn_stats_kernel(const float* __restrict__ wy,
                                                       float* __restrict__ stats) {
  int c = blockIdx.x;
  float s = 0.f, s2 = 0.f;
  for (int i = threadIdx.x; i < BN_ * N_; i += 256) {
    int b = i >> 10;
    int n = i & (N_ - 1);
    float v = wy[((long long)b * C_ + c) * N_ + n];
    s += v; s2 += v * v;
  }
  __shared__ float rs[256], rq[256];
  rs[threadIdx.x] = s; rq[threadIdx.x] = s2;
  __syncthreads();
  for (int st = 128; st > 0; st >>= 1) {
    if (threadIdx.x < st) { rs[threadIdx.x] += rs[threadIdx.x + st]; rq[threadIdx.x] += rq[threadIdx.x + st]; }
    __syncthreads();
  }
  if (threadIdx.x == 0) {
    float inv_n = 1.0f / (BN_ * N_);
    float mean = rs[0] * inv_n;
    float var  = rq[0] * inv_n - mean * mean;
    stats[c] = mean;
    stats[C_ + c] = rsqrtf(var + 1e-5f);
  }
}

__global__ __launch_bounds__(256) void bn_apply_kernel(const float* __restrict__ wy,
                                                       const float* __restrict__ stats,
                                                       const float* __restrict__ gamma,
                                                       const float* __restrict__ beta,
                                                       const float* __restrict__ x,
                                                       float* __restrict__ out) {
  long long i = (long long)blockIdx.x * 256 + threadIdx.x;
  int c = (int)((i >> 10) & (C_ - 1));
  float m = stats[c], inv = stats[C_ + c];
  out[i] = (wy[i] - m) * inv * gamma[c] + beta[c] + x[i];
}

// ---------------- host launcher ----------------
extern "C" void kernel_launch(void* const* d_in, const int* in_sizes, int n_in,
                              void* d_out, int out_size, void* d_ws, size_t ws_size,
                              hipStream_t stream) {
  const float* x    = (const float*)d_in[0];
  const float* xl   = (const float*)d_in[1];
  const float* wg   = (const float*)d_in[2];
  const float* bg   = (const float*)d_in[3];
  const float* wgl  = (const float*)d_in[4];
  const float* bgl  = (const float*)d_in[5];
  const float* wth  = (const float*)d_in[6];
  const float* bth  = (const float*)d_in[7];
  const float* wph  = (const float*)d_in[8];
  const float* bph  = (const float*)d_in[9];
  const float* wphl = (const float*)d_in[10];
  const float* bphl = (const float*)d_in[11];
  const float* wW   = (const float*)d_in[12];
  const float* bW   = (const float*)d_in[13];
  const float* gam  = (const float*)d_in[14];
  const float* bet  = (const float*)d_in[15];
  float* out = (float*)d_out;

  char* ws = (char*)d_ws;
  size_t off = 0;
  auto take = [&](size_t bytes) { char* p = ws + off; off += (bytes + 255) & ~(size_t)255; return p; };

  __bf16* xfbf  = (__bf16*)take(2ull * BN_ * N_  * C_);
  __bf16* xlfbf = (__bf16*)take(2ull * BN_ * NL_ * IC_);
  __bf16* wgbf  = (__bf16*)take(2ull * IC_ * C_);
  __bf16* wglbf = (__bf16*)take(2ull * IC_ * IC_);
  __bf16* wthbf = (__bf16*)take(2ull * IC_ * C_);
  __bf16* wphbf = (__bf16*)take(2ull * IC_ * C_);
  __bf16* wphlbf= (__bf16*)take(2ull * IC_ * IC_);
  __bf16* wWbf  = (__bf16*)take(2ull * C_ * IC_);
  __bf16* Qbf   = (__bf16*)take(2ull * BN_ * N_ * IC_);
  __bf16* KVbf  = (__bf16*)take(2ull * BN_ * M_ * IC_);
  __bf16* Vtbf  = (__bf16*)take(2ull * BN_ * IC_ * M_);
  float*  stats = (float*)take(4ull * 2 * C_);
  __bf16* Ybf = xfbf;          // reuse: xfbf dead after projection GEMMs
  float*  wyf = (float*)xlfbf; // reuse: xlfbf dead after projection GEMMs
  (void)ws_size; (void)in_sizes; (void)n_in; (void)out_size;

  transpose_cvt_kernel<<<(BN_ * (long long)N_ * C_) / 256, 256, 0, stream>>>(x, xfbf, C_, N_);
  transpose_cvt_kernel<<<(BN_ * (long long)NL_ * IC_) / 256, 256, 0, stream>>>(xl, xlfbf, IC_, NL_);
  cvt_kernel<<<(IC_ * C_) / 256, 256, 0, stream>>>(wg, wgbf, IC_ * C_);
  cvt_kernel<<<(IC_ * IC_) / 256, 256, 0, stream>>>(wgl, wglbf, IC_ * IC_);
  cvt_kernel<<<(IC_ * C_) / 256, 256, 0, stream>>>(wth, wthbf, IC_ * C_);
  cvt_kernel<<<(IC_ * C_) / 256, 256, 0, stream>>>(wph, wphbf, IC_ * C_);
  cvt_kernel<<<(IC_ * IC_) / 256, 256, 0, stream>>>(wphl, wphlbf, IC_ * IC_);
  cvt_kernel<<<(C_ * IC_) / 256, 256, 0, stream>>>(wW, wWbf, C_ * IC_);

  dim3 blk(128);
  gemm_bf16_kernel<<<dim3(N_/16, IC_/256, BN_), blk, 0, stream>>>(
      xfbf, C_, (long long)N_ * C_, wthbf, C_, bth, Qbf, IC_, (long long)N_ * IC_, C_, 0);
  gemm_bf16_kernel<<<dim3(N_/16, IC_/256, BN_), blk, 0, stream>>>(
      xfbf, C_, (long long)N_ * C_, wphbf, C_, bph, KVbf, IC_, (long long)M_ * IC_, C_, 0);
  gemm_bf16_kernel<<<dim3(NL_/16, IC_/256, BN_), blk, 0, stream>>>(
      xlfbf, IC_, (long long)NL_ * IC_, wphlbf, IC_, bphl,
      KVbf + (long long)N_ * IC_, IC_, (long long)M_ * IC_, IC_, 0);
  gemm_bf16_kernel<<<dim3(N_/16, IC_/256, BN_), blk, 0, stream>>>(
      xfbf, C_, (long long)N_ * C_, wgbf, C_, bg, Vtbf, M_, (long long)IC_ * M_, C_, 1);
  gemm_bf16_kernel<<<dim3(NL_/16, IC_/256, BN_), blk, 0, stream>>>(
      xlfbf, IC_, (long long)NL_ * IC_, wglbf, IC_, bgl,
      Vtbf + N_, M_, (long long)IC_ * M_, IC_, 1);

  attn_kernel<<<dim3(N_/16, BN_), blk, 0, stream>>>(Qbf, KVbf, Vtbf, Ybf);

  gemm_bf16_kernel<<<dim3(N_/16, C_/256, BN_), blk, 0, stream>>>(
      Ybf, IC_, (long long)N_ * IC_, wWbf, IC_, bW, wyf, N_, (long long)C_ * N_, IC_, 2);

  bn_stats_kernel<<<C_, 256, 0, stream>>>(wyf, stats);
  bn_apply_kernel<<<(BN_ * (long long)C_ * N_) / 256, 256, 0, stream>>>(wyf, stats, gam, bet, x, out);
}
